// DotProductAttention_2095944040804
// MI455X (gfx1250) — matrix operand order repaired and checked by reading, not compiled, not run on hardware
//
#include <hip/hip_runtime.h>
#include <math.h>

// ---------------------------------------------------------------------------
// Flash attention, B=32, S=2048, D=64 fp32, per-batch key-length masking.
// Both GEMMs on v_wmma_f32_16x16x32_f16 (f16 operands, f32 accumulate).
//
// Structure:
//  * scores computed TRANSPOSED (S^T = K * Q^T): C-layout puts the q-row in
//    the lane index -> softmax stats are per-lane scalars (VALU max3/add
//    trees + one v_permlanex16 cross-half exchange).
//  * whole online softmax runs in the log2 domain: log2(e)/sqrt(d) is folded
//    into Q once, probabilities use raw v_exp_f32 (exp2) with no per-element
//    base-conversion multiply.
//  * P^T is already in B-operand orientation; the off-half key groups are
//    swapped in registers via v_permlanex16 (no LDS round-trip, no barrier).
//  * O accumulates transposed: contiguous 8-float stores per lane.
//  * K and V converted to f16 in LDS once per block (8 waves share), padded
//    rows for conflict-free ds_load_b128; masking peeled to the last chunk
//    (uniform branch -> compiler emits a mask-free steady-state loop).
// ---------------------------------------------------------------------------

typedef __attribute__((ext_vector_type(16))) _Float16 v16h;
typedef __attribute__((ext_vector_type(8)))  _Float16 v8h;
typedef __attribute__((ext_vector_type(8)))  float    v8f;
typedef __attribute__((ext_vector_type(4)))  int      v4i;
typedef __attribute__((ext_vector_type(8)))  int      v8i;

#define BATCH   32
#define SEQ     2048
#define DHEAD   64
#define BLOCK_M 128          // q rows per workgroup (8 waves x 16 rows)
#define CHUNK   32           // keys per iteration
#define KPAD    72           // halves per Kh row (64 data + 8 pad) -> 144B rows
#define VPAD    40           // halves per VT row (32 data + 8 pad) -> 80B rows

#define LOG2E      1.44269504088896340736f
// d2l mask constant (-1e6) expressed in the log2 score domain
#define MASK_NEG2  (-1.0e6f * LOG2E)

// lane <-> lane^16 exchange: VALU v_permlanex16 (ds_bpermute fallback)
__device__ __forceinline__ int xor16_i32(int x) {
#if __has_builtin(__builtin_amdgcn_permlanex16)
    return __builtin_amdgcn_permlanex16(x, x, 0x76543210, 0xfedcba98, false, false);
#else
    return __shfl_xor(x, 16, 32);
#endif
}
__device__ __forceinline__ float xor16_f32(float x) {
    return __builtin_bit_cast(float, xor16_i32(__builtin_bit_cast(int, x)));
}

__global__ __launch_bounds__(256, 2)
void fa_fwd_kernel(const float* __restrict__ Q,
                   const float* __restrict__ K,
                   const float* __restrict__ V,
                   const int*   __restrict__ VL,
                   float*       __restrict__ O)
{
    __shared__ __attribute__((aligned(16))) _Float16 Kh[CHUNK][KPAD];  // K chunk, f16 row-major
    __shared__ __attribute__((aligned(16))) _Float16 VT[DHEAD][VPAD];  // V chunk, f16 transposed

    const int tid  = threadIdx.x;
    const int lane = tid & 31;
    const int wave = tid >> 5;
    const int half = lane >> 4;
    const int l16  = lane & 15;

    const int b      = blockIdx.y;
    const int q_base = blockIdx.x * BLOCK_M + wave * 16;

    const float* Qb = Q + ((size_t)b * SEQ + q_base) * DHEAD;
    const float* Kb = K + (size_t)b * SEQ * DHEAD;
    const float* Vb = V + (size_t)b * SEQ * DHEAD;

    const int vl     = VL[b];                    // [1, SEQ], uniform per block
    const int nchunk = (vl + CHUNK - 1) >> 5;    // early exit past valid keys

    // ---- loop-invariant B operand: Q^T with log2(e)/sqrt(64) folded in.
    // B(32x16) f16, this code's contiguous order (A matches): lane(half, n=l16), element e -> d = half*16 + e
    const float qscale = 0.125f * LOG2E;
    v16h bQ[2];
    {
        const float* qr = Qb + (size_t)l16 * DHEAD;
        #pragma unroll
        for (int dc = 0; dc < 2; ++dc)
            #pragma unroll
            for (int e = 0; e < 16; ++e)
                bQ[dc][e] = (_Float16)(qr[dc * 32 + half * 16 + e] * qscale);
    }

    // ---- state: per-lane q-row stats (log2 domain); O^T accumulators.
    float mrow = -3.0e38f, lrow = 0.0f;
    v8f acc[4];
    #pragma unroll
    for (int dt = 0; dt < 4; ++dt)
        #pragma unroll
        for (int r = 0; r < 8; ++r) acc[dt][r] = 0.0f;

    for (int c = 0; c < nchunk; ++c) {
        const int  key0 = c << 5;
        const bool last = (c + 1 == nchunk);     // uniform branch

        // ---- cooperative staging: K row-major f16, V transposed f16
        __syncthreads();
        {
            #pragma unroll
            for (int it = 0; it < 4; ++it) {     // 1024 f16-pairs over 256 thr
                const int pp = tid + it * 256;
                const int k  = pp >> 5;
                const int d  = (pp & 31) << 1;
                const float* kr = Kb + (size_t)(key0 + k) * DHEAD + d;
                Kh[k][d]     = (_Float16)kr[0];
                Kh[k][d + 1] = (_Float16)kr[1];
            }
            const int dv = tid & 63;
            const int kb = (tid >> 6) << 1;
            #pragma unroll
            for (int it = 0; it < 4; ++it) {
                const int k = kb + it * 8;
                VT[dv][k]     = (_Float16)Vb[(size_t)(key0 + k)     * DHEAD + dv];
                VT[dv][k + 1] = (_Float16)Vb[(size_t)(key0 + k + 1) * DHEAD + dv];
            }
            if (key0 + CHUNK < SEQ) {            // nudge next chunk toward L2
                __builtin_prefetch(Kb + (size_t)(key0 + CHUNK + (tid >> 3)) * DHEAD, 0, 0);
                __builtin_prefetch(Vb + (size_t)(key0 + CHUNK + (tid >> 3)) * DHEAD, 0, 0);
            }
        }
        __syncthreads();

        // ---- S^T tiles: A = K rows (16 keys x 32 d), B = Q^T. C: lane = q.
        v8f s[2];
        #pragma unroll
        for (int t = 0; t < 2; ++t) {
            v16h aK0, aK1;
            const _Float16* krow = &Kh[t * 16 + l16][0];
            #pragma unroll
            for (int e = 0; e < 16; ++e) {
                const int doff = half * 16 + e;
                aK0[e] = krow[doff];           // d-chunk 0: two 16B ds reads
                aK1[e] = krow[32 + doff];      // d-chunk 1
            }
            v8f cc;
            #pragma unroll
            for (int r = 0; r < 8; ++r) cc[r] = 0.0f;
            cc = __builtin_amdgcn_wmma_f32_16x16x32_f16(
                     false, aK0, false, bQ[0], (short)0, cc, false, false);
            cc = __builtin_amdgcn_wmma_f32_16x16x32_f16(
                     false, aK1, false, bQ[1], (short)0, cc, false, false);
            s[t] = cc;
        }

        // ---- mask only on the final chunk (keys run along VGPR index)
        if (last) {
            #pragma unroll
            for (int t = 0; t < 2; ++t) {
                const int kbase = key0 + t * 16 + 8 * half;
                #pragma unroll
                for (int r = 0; r < 8; ++r)
                    s[t][r] = (kbase + r < vl) ? s[t][r] : MASK_NEG2;
            }
        }

        // ---- online softmax in log2 domain: max3 tree + one permlanex16
        float mloc = fmaxf(s[0][0], s[0][1]);
        #pragma unroll
        for (int r = 2; r < 8; ++r) mloc = fmaxf(mloc, s[0][r]);
        #pragma unroll
        for (int r = 0; r < 8; ++r) mloc = fmaxf(mloc, s[1][r]);
        mloc = fmaxf(mloc, xor16_f32(mloc));
        const float mnew = fmaxf(mrow, mloc);
        const float corr = __builtin_amdgcn_exp2f(mrow - mnew); // 0 on first chunk
        mrow = mnew;

        float rs = 0.0f;
        #pragma unroll
        for (int t = 0; t < 2; ++t)
            #pragma unroll
            for (int r = 0; r < 8; ++r) {
                const float p = __builtin_amdgcn_exp2f(s[t][r] - mnew);
                s[t][r] = p;
                rs += p;
            }
        rs += xor16_f32(rs);
        lrow = fmaf(lrow, corr, rs);
        #pragma unroll
        for (int dt = 0; dt < 4; ++dt)
            #pragma unroll
            for (int r = 0; r < 8; ++r) acc[dt][r] *= corr;

        // ---- build B operand P^T (32 keys x 16 q) purely in registers:
        // pack to f16 pairs, swap the off-half key groups via permlanex16.
        v8h ph0, ph1;
        #pragma unroll
        for (int r = 0; r < 8; ++r) {
            ph0[r] = (_Float16)s[0][r];
            ph1[r] = (_Float16)s[1][r];
        }
        const v4i p0 = __builtin_bit_cast(v4i, ph0);
        const v4i p1 = __builtin_bit_cast(v4i, ph1);
        v8i bi;
        #pragma unroll
        for (int j = 0; j < 4; ++j) {
            const int x0 = xor16_i32(p0[j]);
            const int x1 = xor16_i32(p1[j]);
            bi[j]     = half ? x1    : p0[j];  // keys half*16 + 0..7
            bi[4 + j] = half ? p1[j] : x0;     // keys half*16 + 8..15
        }
        const v16h bP = __builtin_bit_cast(v16h, bi);

        // ---- O^T += V^T(16d x 32k) * P^T(32k x 16q), one K=32 WMMA per d tile
        #pragma unroll
        for (int dt = 0; dt < 4; ++dt) {
            v16h av;
            const _Float16* vrow = &VT[dt * 16 + l16][0];
            #pragma unroll
            for (int e = 0; e < 16; ++e)
                av[e] = vrow[half * 16 + e];
            acc[dt] = __builtin_amdgcn_wmma_f32_16x16x32_f16(
                          false, av, false, bP, (short)0, acc[dt], false, false);
        }
    }

    // ---- epilogue: O^T C-layout -> contiguous 8-float stores per lane
    const float inv = 1.0f / lrow;
    const int   q   = q_base + l16;
    float* orow = O + ((size_t)b * SEQ + q) * DHEAD;
    #pragma unroll
    for (int dt = 0; dt < 4; ++dt)
        #pragma unroll
        for (int r = 0; r < 8; ++r)
            orow[dt * 16 + 8 * half + r] = acc[dt][r] * inv;
}

extern "C" void kernel_launch(void* const* d_in, const int* in_sizes, int n_in,
                              void* d_out, int out_size, void* d_ws, size_t ws_size,
                              hipStream_t stream) {
    (void)in_sizes; (void)n_in; (void)out_size; (void)d_ws; (void)ws_size;
    const float* Q  = (const float*)d_in[0];
    const float* K  = (const float*)d_in[1];
    const float* V  = (const float*)d_in[2];
    const int*   VL = (const int*)d_in[3];
    float*       O  = (float*)d_out;

    dim3 grid(SEQ / BLOCK_M, BATCH, 1);
    dim3 block(256, 1, 1);
    fa_fwd_kernel<<<grid, block, 0, stream>>>(Q, K, V, VL, O);
}
